// GroupQueryAttention_55542517072601
// MI455X (gfx1250) — matrix operand order repaired and checked by reading, not compiled, not run on hardware
//
#include <hip/hip_runtime.h>
#include <hip/hip_bf16.h>

// ---------------- problem constants ----------------
constexpr int kE  = 2048;
constexpr int kNH = 16;
constexpr int kG  = 4;
constexpr int kHD = kE / kNH;   // 128
constexpr int kKV = kE / kG;    // 512
constexpr int kB  = 2;
constexpr int kS  = 2048;
constexpr int kM  = kB * kS;    // 4096 rows for all projections

typedef __attribute__((ext_vector_type(16))) _Float16     v16h;
typedef __attribute__((ext_vector_type(8)))  _Float16     v8h;
typedef __attribute__((ext_vector_type(8)))  float        v8f;
typedef __attribute__((ext_vector_type(4)))  unsigned int u32x4;
typedef __attribute__((ext_vector_type(8)))  unsigned int u32x8;

// ---------------- WMMA fragment helpers (wave32, 16x16x32 f16) ----------------
// A (16Mx32K): lanes 0-15 -> row M=lane, K {kb..kb+7} U {kb+16..kb+23}, kb=0
//              lanes 16-31 -> row M=lane-16, same with kb=8
__device__ __forceinline__ v16h load_a_f16(const _Float16* base, int ld, int m0, int k0, int lane) {
  int m  = m0 + (lane & 15);
  int kb = k0 + ((lane >> 4) << 3);
  const _Float16* p = base + (size_t)m * ld + kb;
  v8h lo = *(const v8h*)(p);
  v8h hi = *(const v8h*)(p + 16);
  v16h a;
#pragma unroll
  for (int i = 0; i < 8; ++i) { a[i] = lo[i]; a[8 + i] = hi[i]; }
  return a;
}

__device__ __forceinline__ v16h load_a_f32(const float* base, int ld, int m0, int k0, int lane) {
  int m  = m0 + (lane & 15);
  int kb = k0 + ((lane >> 4) << 3);
  const float* p = base + (size_t)m * ld + kb;
  v16h a;
#pragma unroll
  for (int i = 0; i < 8; ++i) a[i] = (_Float16)p[i];
#pragma unroll
  for (int i = 0; i < 8; ++i) a[8 + i] = (_Float16)p[16 + i];
  return a;
}

// B (32Kx16N): lanes 0-15 -> col N=lane, K {0..7} U {16..23} ; lanes 16-31 -> col N=lane-16, K {8..15} U {24..31}
// Source stored "transposed": Bt[n][k] row-major, leading dim ld -> two 16B loads per lane (k offsets kb and kb+16).
__device__ __forceinline__ v16h load_b_f16(const _Float16* base, int ld, int n0, int k0, int lane) {
  int n  = n0 + (lane & 15);
  int kb = k0 + ((lane >> 4) << 3);
  const _Float16* p = base + (size_t)n * ld + kb;
  v8h lo = *(const v8h*)(p);
  v8h hi = *(const v8h*)(p + 16);
  v16h b;
#pragma unroll
  for (int i = 0; i < 8; ++i) { b[i] = lo[i]; b[8 + i] = hi[i]; }
  return b;
}

__device__ __forceinline__ v8f wmma_f16(v16h a, v16h b, v8f c) {
  return __builtin_amdgcn_wmma_f32_16x16x32_f16(false, a, false, b, (short)0, c, false, false);
}

__device__ __forceinline__ float half_reduce_max(float v) {
#pragma unroll
  for (int m = 1; m < 16; m <<= 1) v = fmaxf(v, __shfl_xor(v, m, 32));
  return v;
}
__device__ __forceinline__ float half_reduce_sum(float v) {
#pragma unroll
  for (int m = 1; m < 16; m <<= 1) v += __shfl_xor(v, m, 32);
  return v;
}

// Low 32 bits of a generic pointer to LDS == LDS byte offset (aperture layout).
// Laundered through asm so the addrspacecast can never be folded into a static initializer.
__device__ __forceinline__ unsigned lds_lo32(const void* p) {
  unsigned v = (unsigned)(uintptr_t)p;
  unsigned out;
  asm("v_mov_b32 %0, %1" : "=v"(out) : "v"(v));
  return out;
}

// ---------------- Tensor Data Mover: 2D tile global->LDS (D# groups 0+1) ----------------
// data_size = 2 bytes. Strides/dims in elements. Groups 2/3 omitted (<=2D tensor).
__device__ __forceinline__ void tdm_load_2d(unsigned lds_off, const void* gptr,
                                            unsigned tile_d0, unsigned tile_d1,
                                            unsigned long long stride0,
                                            unsigned tensor_d0, unsigned tensor_d1) {
  unsigned long long ga = (unsigned long long)(uintptr_t)gptr;
  u32x4 g0;
  g0[0] = 1u;                                              // count=1, user descriptor
  g0[1] = lds_off;                                         // LDS byte address
  g0[2] = (unsigned)ga;                                    // global_addr[31:0]
  g0[3] = (unsigned)((ga >> 32) & 0x1FFFFFFu) | (2u << 30);// global_addr[56:32] | type=2
  u32x8 g1;
  g1[0] = (1u << 16);                                      // data_size=1 (2 bytes)
  g1[1] = (tensor_d0 & 0xFFFFu) << 16;                     // tensor_dim0[15:0]
  g1[2] = (tensor_d0 >> 16) | ((tensor_d1 & 0xFFFFu) << 16);
  g1[3] = (tensor_d1 >> 16) | (tile_d0 << 16);             // tile_dim0
  g1[4] = tile_d1 & 0xFFFFu;                               // tile_dim1 (tile_dim2=0)
  g1[5] = (unsigned)(stride0 & 0xFFFFFFFFull);             // tensor_dim0_stride[31:0]
  g1[6] = (unsigned)((stride0 >> 32) & 0xFFFFull);         // stride[47:32] | s1[15:0]=0
  g1[7] = 0u;                                              // tensor_dim1_stride hi = 0
  asm volatile("tensor_load_to_lds %0, %1" :: "s"(g0), "s"(g1) : "memory");
}

// ---------------- kernel 1: W[K][N] f32 -> Wt[N][K] f16 ----------------
__global__ void __launch_bounds__(256)
transpose_cvt_kernel(const float* __restrict__ W, _Float16* __restrict__ Wt, int K, int N) {
  __shared__ float tile[16][17];
  int tx = threadIdx.x, ty = threadIdx.y;
  int n0 = blockIdx.x * 16, k0 = blockIdx.y * 16;
  tile[ty][tx] = W[(size_t)(k0 + ty) * N + (n0 + tx)];
  __syncthreads();
  Wt[(size_t)(n0 + ty) * K + (k0 + tx)] = (_Float16)tile[tx][ty];
}

// ---------------- kernel 2: Y(f16) = X(f32) @ Wt^T + bias ; wave = 32x64 tile, ping-pong K ----
__global__ void __launch_bounds__(32)
proj_gemm_kernel(const float* __restrict__ X, const _Float16* __restrict__ Wt,
                 const float* __restrict__ bias, _Float16* __restrict__ Y,
                 int N, int K, int transposed) {
  int lane = threadIdx.x;
  int n0 = blockIdx.x * 64;
  int m0 = blockIdx.y * 32;
  v8f acc[2][4] = {};
  v16h aA[2], bA[4], aB[2], bB[4];
  aA[0] = load_a_f32(X, K, m0,      0, lane);
  aA[1] = load_a_f32(X, K, m0 + 16, 0, lane);
#pragma unroll
  for (int t = 0; t < 4; ++t) bA[t] = load_b_f16(Wt, K, n0 + 16 * t, 0, lane);

  for (int k = 0; k < K; k += 64) {
    // prefetch half B while computing half A
    aB[0] = load_a_f32(X, K, m0,      k + 32, lane);
    aB[1] = load_a_f32(X, K, m0 + 16, k + 32, lane);
#pragma unroll
    for (int t = 0; t < 4; ++t) bB[t] = load_b_f16(Wt, K, n0 + 16 * t, k + 32, lane);
#pragma unroll
    for (int t = 0; t < 4; ++t) {
      acc[0][t] = wmma_f16(aA[0], bA[t], acc[0][t]);
      acc[1][t] = wmma_f16(aA[1], bA[t], acc[1][t]);
    }
    if (k + 64 < K) {
      aA[0] = load_a_f32(X, K, m0,      k + 64, lane);
      aA[1] = load_a_f32(X, K, m0 + 16, k + 64, lane);
#pragma unroll
      for (int t = 0; t < 4; ++t) bA[t] = load_b_f16(Wt, K, n0 + 16 * t, k + 64, lane);
    }
#pragma unroll
    for (int t = 0; t < 4; ++t) {
      acc[0][t] = wmma_f16(aB[0], bB[t], acc[0][t]);
      acc[1][t] = wmma_f16(aB[1], bB[t], acc[1][t]);
    }
  }

  int hi = lane >> 4, nc = lane & 15;
#pragma unroll
  for (int mi = 0; mi < 2; ++mi) {
    int mb = m0 + 16 * mi;
    if (!transposed) {
#pragma unroll
      for (int t = 0; t < 4; ++t) {
        int n = n0 + 16 * t + nc;
        float bv = bias[n];
#pragma unroll
        for (int r = 0; r < 8; ++r)
          Y[(size_t)(mb + r + 8 * hi) * N + n] = (_Float16)(acc[mi][t][r] + bv);
      }
    } else {
      // Y is Vt[kB][N][kS]; 8 consecutive s -> packed 16B store
      int b  = mb / kS;
      int s0 = (mb % kS) + 8 * hi;
#pragma unroll
      for (int t = 0; t < 4; ++t) {
        int n = n0 + 16 * t + nc;
        float bv = bias[n];
        v8h pack;
#pragma unroll
        for (int r = 0; r < 8; ++r) pack[r] = (_Float16)(acc[mi][t][r] + bv);
        *(v8h*)(Y + ((size_t)b * N + n) * kS + s0) = pack;
      }
    }
  }
}

// ---------------- kernel 3: GQA flash attention ; wave = (b, head, 16 query rows) ----------------
// K/V 32-key chunks staged in LDS by the Tensor Data Mover, double buffered on TENSORcnt.
__global__ void __launch_bounds__(32)
gqa_attn_kernel(const _Float16* __restrict__ Qh, const _Float16* __restrict__ Kh,
                const _Float16* __restrict__ Vt, _Float16* __restrict__ AO) {
  __shared__ __align__(16) _Float16 sm[2 * 32 * kHD + 2 * kHD * 32 + 16 * 32];
  _Float16* kbuf0 = sm;
  _Float16* kbuf1 = sm + 32 * kHD;
  _Float16* vbuf0 = sm + 2 * 32 * kHD;
  _Float16* vbuf1 = sm + 2 * 32 * kHD + kHD * 32;
  _Float16* pbuf  = sm + 2 * 32 * kHD + 2 * kHD * 32;
  const unsigned kBufBytes = 32 * kHD * 2;  // 8 KB per K/V buffer
  unsigned sm_base = lds_lo32(sm);          // LDS byte offset of sm (runtime, not foldable)

  int lane = threadIdx.x;
  int m0 = blockIdx.x * 16;
  int h  = blockIdx.y;
  int b  = blockIdx.z;
  int g  = h >> 2;                           // NH/G = 4 query heads per kv head
  const float scale = 0.088388347648318447f; // 1/sqrt(128)

  const _Float16* qbase = Qh + (size_t)b * kS * kE + (size_t)h * kHD;     // ld = kE
  const _Float16* kbase = Kh + (size_t)b * kS * kKV + (size_t)g * kHD;    // ld = kKV
  const _Float16* vbase = Vt + ((size_t)b * kKV + (size_t)g * kHD) * kS;  // ld = kS

  v16h qf[4];
#pragma unroll
  for (int st = 0; st < 4; ++st) qf[st] = load_a_f16(qbase, kE, m0, 32 * st, lane);

  v8f acc[8] = {};
  float mrow[8], lsum[8];
#pragma unroll
  for (int r = 0; r < 8; ++r) { mrow[r] = -1e30f; lsum[r] = 0.0f; }
  int hi = lane >> 4, nc = lane & 15;

  // preload chunk 0: K tile 32 keys x 128 hd (row stride kKV); V tile 128 hd x 32 keys (row stride kS)
  tdm_load_2d(sm_base,                 kbase, kHD, 32, (unsigned long long)kKV, kHD, 32);
  tdm_load_2d(sm_base + 2 * kBufBytes, vbase, 32, kHD, (unsigned long long)kS,  32, kHD);

  for (int kc = 0, it = 0; kc < kS; kc += 32, ++it) {
    int p = it & 1;
    __builtin_amdgcn_s_wait_tensorcnt(0);  // chunk `it` resident in LDS
    __syncthreads();                        // ordering: prior DS reads done before TDM reuses buffers
    if (kc + 32 < kS) {                     // DMA chunk it+1 into the other buffers
      unsigned koff_n = sm_base + (p ? 0u : kBufBytes);
      unsigned voff_n = sm_base + 2 * kBufBytes + (p ? 0u : kBufBytes);
      tdm_load_2d(koff_n, kbase + (size_t)(kc + 32) * kKV, kHD, 32, (unsigned long long)kKV, kHD, 32);
      tdm_load_2d(voff_n, vbase + (kc + 32),               32, kHD, (unsigned long long)kS,  32, kHD);
    }
    const _Float16* kb = p ? kbuf1 : kbuf0;
    const _Float16* vb = p ? vbuf1 : vbuf0;

    // scores: two 16x16 tiles over the 32 keys, K-dim = HD (4 wmma steps), B frags from LDS
    v8f c0 = {}, c1 = {};
#pragma unroll
    for (int st = 0; st < 4; ++st) {
      v16h b0 = load_b_f16(kb, kHD, 0,  32 * st, lane);
      v16h b1 = load_b_f16(kb, kHD, 16, 32 * st, lane);
      c0 = wmma_f16(qf[st], b0, c0);
      c1 = wmma_f16(qf[st], b1, c1);
    }
    // online softmax (row m = r + 8*hi lives in one half-wave per VGPR r)
    float cf[8];
    v8f p0, p1;
#pragma unroll
    for (int r = 0; r < 8; ++r) {
      float s0 = c0[r] * scale, s1 = c1[r] * scale;
      float tm = half_reduce_max(fmaxf(s0, s1));
      float nm = fmaxf(mrow[r], tm);
      float corr = __expf(mrow[r] - nm);
      float e0 = __expf(s0 - nm), e1 = __expf(s1 - nm);
      float ts = half_reduce_sum(e0 + e1);
      lsum[r] = lsum[r] * corr + ts;
      mrow[r] = nm;
      cf[r] = corr;
      p0[r] = e0; p1[r] = e1;
    }
#pragma unroll
    for (int t = 0; t < 8; ++t)
#pragma unroll
      for (int r = 0; r < 8; ++r) acc[t][r] *= cf[r];

    // C-layout -> A-layout transpose of P through LDS
    __syncthreads();
#pragma unroll
    for (int r = 0; r < 8; ++r) {
      int m = r + 8 * hi;
      pbuf[m * 32 + nc]      = (_Float16)p0[r];
      pbuf[m * 32 + 16 + nc] = (_Float16)p1[r];
    }
    __syncthreads();
    v16h pa = load_a_f16(pbuf, 32, 0, 0, lane);

    // out += P(16x32) @ V(32x128), 8 N-tiles of HD, B frags from LDS
#pragma unroll
    for (int t = 0; t < 8; ++t) {
      v16h vfrag = load_b_f16(vb, 32, 16 * t, 0, lane);
      acc[t] = wmma_f16(pa, vfrag, acc[t]);
    }
  }

  _Float16* outp = AO + ((size_t)b * kS + m0) * kE + (size_t)h * kHD;
#pragma unroll
  for (int t = 0; t < 8; ++t) {
    int n = 16 * t + nc;
#pragma unroll
    for (int r = 0; r < 8; ++r)
      outp[(size_t)(r + 8 * hi) * kE + n] = (_Float16)(acc[t][r] / lsum[r]);
  }
}

// ---------------- kernel 4: out(f32) = AO(f16) @ Wt_o^T + bo ; wave = 32x64, ping-pong ----------
__global__ void __launch_bounds__(32)
out_gemm_kernel(const _Float16* __restrict__ A, const _Float16* __restrict__ Wt,
                const float* __restrict__ bias, float* __restrict__ Y, int N, int K) {
  int lane = threadIdx.x;
  int n0 = blockIdx.x * 64;
  int m0 = blockIdx.y * 32;
  v8f acc[2][4] = {};
  v16h aA[2], bA[4], aB[2], bB[4];
  aA[0] = load_a_f16(A, K, m0,      0, lane);
  aA[1] = load_a_f16(A, K, m0 + 16, 0, lane);
#pragma unroll
  for (int t = 0; t < 4; ++t) bA[t] = load_b_f16(Wt, K, n0 + 16 * t, 0, lane);

  for (int k = 0; k < K; k += 64) {
    aB[0] = load_a_f16(A, K, m0,      k + 32, lane);
    aB[1] = load_a_f16(A, K, m0 + 16, k + 32, lane);
#pragma unroll
    for (int t = 0; t < 4; ++t) bB[t] = load_b_f16(Wt, K, n0 + 16 * t, k + 32, lane);
#pragma unroll
    for (int t = 0; t < 4; ++t) {
      acc[0][t] = wmma_f16(aA[0], bA[t], acc[0][t]);
      acc[1][t] = wmma_f16(aA[1], bA[t], acc[1][t]);
    }
    if (k + 64 < K) {
      aA[0] = load_a_f16(A, K, m0,      k + 64, lane);
      aA[1] = load_a_f16(A, K, m0 + 16, k + 64, lane);
#pragma unroll
      for (int t = 0; t < 4; ++t) bA[t] = load_b_f16(Wt, K, n0 + 16 * t, k + 64, lane);
    }
#pragma unroll
    for (int t = 0; t < 4; ++t) {
      acc[0][t] = wmma_f16(aB[0], bB[t], acc[0][t]);
      acc[1][t] = wmma_f16(aB[1], bB[t], acc[1][t]);
    }
  }

  int hi = lane >> 4, nc = lane & 15;
#pragma unroll
  for (int mi = 0; mi < 2; ++mi) {
    int mb = m0 + 16 * mi;
#pragma unroll
    for (int t = 0; t < 4; ++t) {
      int n = n0 + 16 * t + nc;
      float bv = bias[n];
#pragma unroll
      for (int r = 0; r < 8; ++r)
        Y[(size_t)(mb + r + 8 * hi) * N + n] = acc[mi][t][r] + bv;
    }
  }
}

// ---------------- host launcher ----------------
extern "C" void kernel_launch(void* const* d_in, const int* in_sizes, int n_in,
                              void* d_out, int out_size, void* d_ws, size_t ws_size,
                              hipStream_t stream) {
  const float* query  = (const float*)d_in[0];
  const float* key_in = (const float*)d_in[1];
  const float* value  = (const float*)d_in[2];
  const float* Wq = (const float*)d_in[3];
  const float* bq = (const float*)d_in[4];
  const float* Wk = (const float*)d_in[5];
  const float* bk = (const float*)d_in[6];
  const float* Wv = (const float*)d_in[7];
  const float* bv = (const float*)d_in[8];
  const float* Wo = (const float*)d_in[9];
  const float* bo = (const float*)d_in[10];
  float* out = (float*)d_out;

  // workspace carve-up (f16 buffers), ~60 MB total
  char* ws = (char*)d_ws;
  _Float16* Wt_q = (_Float16*)ws;  ws += (size_t)kE * kE * sizeof(_Float16);
  _Float16* Wt_k = (_Float16*)ws;  ws += (size_t)kKV * kE * sizeof(_Float16);
  _Float16* Wt_v = (_Float16*)ws;  ws += (size_t)kKV * kE * sizeof(_Float16);
  _Float16* Wt_o = (_Float16*)ws;  ws += (size_t)kE * kE * sizeof(_Float16);
  _Float16* Qh   = (_Float16*)ws;  ws += (size_t)kM * kE * sizeof(_Float16);
  _Float16* Kh   = (_Float16*)ws;  ws += (size_t)kM * kKV * sizeof(_Float16);
  _Float16* Vt   = (_Float16*)ws;  ws += (size_t)kM * kKV * sizeof(_Float16);
  _Float16* AO   = (_Float16*)ws;  ws += (size_t)kM * kE * sizeof(_Float16);

  dim3 tb16(16, 16);
  transpose_cvt_kernel<<<dim3(kE / 16,  kE / 16), tb16, 0, stream>>>(Wq, Wt_q, kE, kE);
  transpose_cvt_kernel<<<dim3(kKV / 16, kE / 16), tb16, 0, stream>>>(Wk, Wt_k, kE, kKV);
  transpose_cvt_kernel<<<dim3(kKV / 16, kE / 16), tb16, 0, stream>>>(Wv, Wt_v, kE, kKV);
  transpose_cvt_kernel<<<dim3(kE / 16,  kE / 16), tb16, 0, stream>>>(Wo, Wt_o, kE, kE);

  proj_gemm_kernel<<<dim3(kE / 64,  kM / 32), 32, 0, stream>>>(query,  Wt_q, bq, Qh, kE,  kE, 0);
  proj_gemm_kernel<<<dim3(kKV / 64, kM / 32), 32, 0, stream>>>(key_in, Wt_k, bk, Kh, kKV, kE, 0);
  proj_gemm_kernel<<<dim3(kKV / 64, kM / 32), 32, 0, stream>>>(value,  Wt_v, bv, Vt, kKV, kE, 1);

  gqa_attn_kernel<<<dim3(kS / 16, kNH, kB), 32, 0, stream>>>(Qh, Kh, Vt, AO);

  out_gemm_kernel<<<dim3(kE / 64, kM / 32), 32, 0, stream>>>(AO, Wt_o, bo, out, kE, kE);
}
